// GPUDirectLinear_79018808312185
// MI455X (gfx1250) — compile-verified, matrix operand order checked by reading, not run on hardware
//
#include <hip/hip_runtime.h>

typedef __attribute__((ext_vector_type(16))) _Float16 v16h;
typedef __attribute__((ext_vector_type(8)))  _Float16 v8h;
typedef __attribute__((ext_vector_type(8)))  float    v8f;
typedef __attribute__((ext_vector_type(4)))  float    v4f;
typedef __attribute__((ext_vector_type(4)))  int      v4i;

namespace {
constexpr int kM = 8192;       // B*S
constexpr int kN = 11008;      // DOUT
constexpr int kK = 4096;       // DIN
constexpr int BM = 256;        // workgroup tile M
constexpr int BN = 128;        // workgroup tile N
constexpr int BK = 32;         // K step (f16 WMMA K)
constexpr int LDSS = BK + 8;   // padded LDS row stride in halfs (80 B rows, 16B-aligned)
constexpr int NK = kK / BK;    // 128 K-steps
}

// Per-thread register staging for one (A,B) K-slab: A 256x32 f32, B 128x32 i32
struct Stage {
    v4f a[8];
    v4i b[4];
};

__device__ __forceinline__ void global_fetch(Stage& st,
                                             const float* __restrict__ X,
                                             const int* __restrict__ Wq,
                                             int block_m, int block_n, int k0,
                                             int ldr, int ldc) {
#pragma unroll
    for (int p = 0; p < 8; ++p)
        st.a[p] = *(const v4f*)(X  + (size_t)(block_m + p * 32 + ldr) * kK + k0 + ldc);
#pragma unroll
    for (int p = 0; p < 4; ++p)
        st.b[p] = *(const v4i*)(Wq + (size_t)(block_n + p * 32 + ldr) * kK + k0 + ldc);
}

__device__ __forceinline__ void lds_stage(const Stage& st,
                                          _Float16* __restrict__ sA,
                                          _Float16* __restrict__ sB,
                                          int ldr, int ldc) {
#pragma unroll
    for (int p = 0; p < 8; ++p) {
        _Float16* da = sA + (p * 32 + ldr) * LDSS + ldc;
        da[0] = (_Float16)st.a[p].x;
        da[1] = (_Float16)st.a[p].y;
        da[2] = (_Float16)st.a[p].z;
        da[3] = (_Float16)st.a[p].w;
    }
#pragma unroll
    for (int p = 0; p < 4; ++p) {
        _Float16* db = sB + (p * 32 + ldr) * LDSS + ldc;
        db[0] = (_Float16)st.b[p].x;   // int8 value -> f16, exact
        db[1] = (_Float16)st.b[p].y;
        db[2] = (_Float16)st.b[p].z;
        db[3] = (_Float16)st.b[p].w;
    }
}

__device__ __forceinline__ v16h frag_load(const _Float16* base) {
    v8h lo = *(const v8h*)(base);        // K = frag_k .. frag_k+7
    v8h hi = *(const v8h*)(base + 16);   // K = frag_k+16 .. frag_k+23
    return __builtin_shufflevector(lo, hi,
        0,1,2,3,4,5,6,7,8,9,10,11,12,13,14,15);
}

__device__ __forceinline__ void wave_mma(const _Float16* __restrict__ sA,
                                         const _Float16* __restrict__ sB,
                                         int wave_m, int wave_n,
                                         int frag_r, int frag_k,
                                         v8f acc[4][4]) {
    v16h af[4];
#pragma unroll
    for (int i = 0; i < 4; ++i)
        af[i] = frag_load(sA + (wave_m + i * 16 + frag_r) * LDSS + frag_k);
#pragma unroll
    for (int j = 0; j < 4; ++j) {
        v16h bf = frag_load(sB + (wave_n + j * 16 + frag_r) * LDSS + frag_k);
#pragma unroll
        for (int i = 0; i < 4; ++i)
            acc[i][j] = __builtin_amdgcn_wmma_f32_16x16x32_f16(
                false, af[i], false, bf, (short)0, acc[i][j], false, false);
    }
}

__global__ __launch_bounds__(256)
void w8a16_wmma_gemm(const float* __restrict__ X,
                     const int* __restrict__ Wq,
                     const float* __restrict__ scale,
                     const float* __restrict__ bias,
                     float* __restrict__ Out)
{
    // ping-pong f16 tiles: A 2x(256x40), B 2x(128x40)  => 60 KB total
    __shared__ _Float16 sA[2][BM * LDSS];
    __shared__ _Float16 sB[2][BN * LDSS];

    const int tid  = threadIdx.x;
    const int lane = tid & 31;
    const int wid  = tid >> 5;           // 8 waves

    const int block_m = blockIdx.x * BM;
    const int block_n = blockIdx.y * BN;

    // wave grid 4 (M) x 2 (N): each wave owns 64x64 = 4x4 tiles of 16x16
    const int wave_m = (wid >> 1) * 64;
    const int wave_n = (wid & 1) * 64;

    v8f acc[4][4] = {};                  // 128 VGPRs of fp32 accumulators

    const int ldr = tid >> 3;            // staging row 0..31 per pass
    const int ldc = (tid & 7) * 4;       // staging dword column

    // WMMA 16-bit fragment addressing (ISA 7.12.2): lane = (k_half<<4) | row
    const int frag_r = lane & 15;
    const int frag_k = (lane >> 4) * 8;

    // ---- prologue: tile 0 ----
    Stage st;
    global_fetch(st, X, Wq, block_m, block_n, 0, ldr, ldc);
    lds_stage(st, sA[0], sB[0], ldr, ldc);
    __syncthreads();

    // ---- main loop: one barrier per iteration, loads hidden under WMMA ----
    for (int k = 0; k < NK - 1; ++k) {
        global_fetch(st, X, Wq, block_m, block_n, (k + 1) * BK, ldr, ldc);
        wave_mma(sA[k & 1], sB[k & 1], wave_m, wave_n, frag_r, frag_k, acc);
        lds_stage(st, sA[(k + 1) & 1], sB[(k + 1) & 1], ldr, ldc);
        __syncthreads();
    }
    wave_mma(sA[(NK - 1) & 1], sB[(NK - 1) & 1], wave_m, wave_n, frag_r, frag_k, acc);

    // ---- epilogue: D layout lane = N (lo/hi half), VGPR r = M row r (+8 hi lanes) ----
    const float s = scale[0];
#pragma unroll
    for (int j = 0; j < 4; ++j) {
        const int n   = block_n + wave_n + j * 16 + (lane & 15);
        const float bv = bias[n];
#pragma unroll
        for (int i = 0; i < 4; ++i) {
            const int m0 = block_m + wave_m + i * 16 + (lane >> 4) * 8;
            float* op = Out + (size_t)m0 * kN + n;
#pragma unroll
            for (int r = 0; r < 8; ++r)
                op[(size_t)r * kN] = acc[i][j][r] * s + bv;
        }
    }
}

extern "C" void kernel_launch(void* const* d_in, const int* in_sizes, int n_in,
                              void* d_out, int out_size, void* d_ws, size_t ws_size,
                              hipStream_t stream) {
    (void)in_sizes; (void)n_in; (void)out_size; (void)d_ws; (void)ws_size;
    const float* x    = (const float*)d_in[0];
    const int*   wq   = (const int*)d_in[1];
    const float* sc   = (const float*)d_in[2];
    const float* bi   = (const float*)d_in[3];
    float* out        = (float*)d_out;
    dim3 grid(kM / BM, kN / BN);   // 32 x 86, exact tiling
    w8a16_wmma_gemm<<<grid, 256, 0, stream>>>(x, wq, sc, bi, out);
}
